// TriangleAttention_12094627906018
// MI455X (gfx1250) — compile-verified, matrix operand order checked by reading, not run on hardware
//
#include <hip/hip_runtime.h>
#include <hip/hip_bf16.h>
#include <stddef.h>

typedef _Float16 half_t;
typedef __attribute__((ext_vector_type(16))) _Float16 v16h;
typedef __attribute__((ext_vector_type(8)))  _Float16 h8;
typedef __attribute__((ext_vector_type(8)))  float    v8f;

#define N_ROWS 65536      // L*L flattened (i,j) rows
#define IN_DIM 128
#define L_DIM  256
#define H_HEADS 4
#define D_HEAD 32

// ---- workspace layout (bytes) ----
#define OFF_W   ((size_t)0)                         // packed f16 weight fragments: 41 tiles * 4 kb * 32 lanes * 16 halves
#define OFF_Q   ((size_t)262144)
#define OFF_K   (OFF_Q + (size_t)N_ROWS*128*2)
#define OFF_V   (OFF_K + (size_t)N_ROWS*128*2)
#define OFF_G   (OFF_V + (size_t)N_ROWS*128*2)
#define OFF_AO  (OFF_G + (size_t)N_ROWS*128*2)
#define OFF_B   (OFF_AO + (size_t)N_ROWS*128*2)     // bias f32 [N][4]

__device__ __forceinline__ v8f wmma32(v16h a, v16h b, v8f c) {
  return __builtin_amdgcn_wmma_f32_16x16x32_f16(false, a, false, b, (short)0, c, false, false);
}

// Build a 16-half fragment from two contiguous 16B chunks (A layout and our
// B layouts both reduce to this: halves 0..7 = chunk0, halves 8..15 = chunk1).
__device__ __forceinline__ v16h frag2(const half_t* c0, const half_t* c1) {
  h8 x0 = *(const h8*)c0;
  h8 x1 = *(const h8*)c1;
  v16h f;
#pragma unroll
  for (int i = 0; i < 8; ++i) { f[i] = x0[i]; f[i + 8] = x1[i]; }
  return f;
}

// Load the 4 K-step B fragments of one 16-col weight tile (independent regs
// so the scheduler can clause the 8 b128 loads and keep WMMAs back-to-back).
__device__ __forceinline__ void loadB4(const half_t* __restrict__ packedW, int t, int lane,
                                       v16h b[4]) {
#pragma unroll
  for (int kb = 0; kb < 4; ++kb)
    b[kb] = *(const v16h*)&packedW[((size_t)(t * 4 + kb) * 32 + lane) * 16];
}

// ---------------- kernel P: pack weights to f16 B-fragments ----------------
// tile index t: 0..23 = w_qkv (384 cols), 24..31 = w_gate, 32 = w_pair (pad to 16), 33..40 = w_out
__global__ __launch_bounds__(128) void kP(const float* __restrict__ wqkv,
                                          const float* __restrict__ wpair,
                                          const float* __restrict__ wgate,
                                          const float* __restrict__ wout,
                                          half_t* __restrict__ packedW) {
  const int t = blockIdx.x;            // 0..40
  const int kb = threadIdx.x >> 5;     // 0..3  (K block of 32)
  const int lane = threadIdx.x & 31;
  const int hi = lane >> 4, n = lane & 15;
  half_t* dst = packedW + ((size_t)(t * 4 + kb) * 32 + lane) * 16;
#pragma unroll
  for (int e = 0; e < 16; ++e) {
    int k = kb * 32 + 8 * hi + (e & 7) + ((e >> 3) * 16);
    float v;
    if (t < 24)       v = wqkv[k * 384 + t * 16 + n];
    else if (t < 32)  v = wgate[k * 128 + (t - 24) * 16 + n];
    else if (t == 32) v = (n < 4) ? wpair[k * 4 + n] : 0.0f;
    else              v = wout[k * 128 + (t - 33) * 16 + n];
    dst[e] = (half_t)v;
  }
}

// ---------------- kernel A: LayerNorm + qkv/gate/bias projections ----------------
__device__ __forceinline__ void storeTileA(int t, v8f acc, int r0, int hi, int ncol,
                                           half_t* __restrict__ qbuf, half_t* __restrict__ kbuf,
                                           half_t* __restrict__ vbuf, half_t* __restrict__ gbuf,
                                           const float* __restrict__ b_gate) {
  if (t < 24) {
    half_t* dst; int cb;
    if (t < 8)       { dst = qbuf; cb = t * 16; }
    else if (t < 16) { dst = kbuf; cb = (t - 8) * 16; }
    else             { dst = vbuf; cb = (t - 16) * 16; }
#pragma unroll
    for (int v = 0; v < 8; ++v)
      dst[(size_t)(r0 + v + 8 * hi) * 128 + cb + ncol] = (half_t)acc[v];
  } else {
    int cb = (t - 24) * 16;
    float bg = b_gate[cb + ncol];
#pragma unroll
    for (int v = 0; v < 8; ++v) {
      float g = __builtin_amdgcn_rcpf(1.0f + __expf(-(acc[v] + bg)));
      gbuf[(size_t)(r0 + v + 8 * hi) * 128 + cb + ncol] = (half_t)g;
    }
  }
}

__global__ __launch_bounds__(128) void kA(const float* __restrict__ z,
                                          const float* __restrict__ gamma,
                                          const float* __restrict__ beta,
                                          const float* __restrict__ b_gate,
                                          const half_t* __restrict__ packedW,
                                          half_t* __restrict__ qbuf, half_t* __restrict__ kbuf,
                                          half_t* __restrict__ vbuf, half_t* __restrict__ gbuf,
                                          float* __restrict__ biasbuf) {
  __shared__ float  zf[4][16 * 128];
  __shared__ half_t zh[4][16 * 128];
  const int w = threadIdx.x >> 5;
  const int lane = threadIdx.x & 31;
  const int r0 = (blockIdx.x * 4 + w) * 16;
  const int m = lane & 15, hi = lane >> 4, ncol = lane & 15;

  const float* zrow = z + (size_t)r0 * 128;
  for (int idx = lane; idx < 2048; idx += 32) zf[w][idx] = zrow[idx];
  __syncthreads();

  // LayerNorm: lane pair (m, m+16) handles row m
  float s = 0.f, sq = 0.f;
  const float* zr = &zf[w][m * 128];
  for (int c = hi * 64; c < hi * 64 + 64; ++c) { float x = zr[c]; s += x; sq += x * x; }
  s  += __shfl_xor(s, 16, 32);
  sq += __shfl_xor(sq, 16, 32);
  float mean = s * (1.0f / 128.0f);
  float var  = sq * (1.0f / 128.0f) - mean * mean;
  float inv  = rsqrtf(var + 1e-5f);
  half_t* zhr = &zh[w][m * 128];
  for (int c = hi * 64; c < hi * 64 + 64; ++c)
    zhr[c] = (half_t)((zr[c] - mean) * inv * gamma[c] + beta[c]);
  __syncthreads();

  // A fragments for the 16x128 zn tile
  v16h afr[4];
#pragma unroll
  for (int kb = 0; kb < 4; ++kb) {
    const half_t* c0 = &zh[w][m * 128 + kb * 32 + hi * 8];
    afr[kb] = frag2(c0, c0 + 16);
  }

  // tiles 0..31 (qkv + gate) as software-pipelined pairs; tile 32 (pair/bias) tail
  v16h b0[4], b1[4];
  loadB4(packedW, 0, lane, b0);
#pragma unroll 1
  for (int tp = 0; tp < 16; ++tp) {
    const int t0 = tp * 2, t1 = tp * 2 + 1;
    loadB4(packedW, t1, lane, b1);
    v8f acc0 = {};
#pragma unroll
    for (int kb = 0; kb < 4; ++kb) acc0 = wmma32(afr[kb], b0[kb], acc0);
    loadB4(packedW, (tp < 15) ? (t1 + 1) : 32, lane, b0);  // prefetch next / tail
    v8f acc1 = {};
#pragma unroll
    for (int kb = 0; kb < 4; ++kb) acc1 = wmma32(afr[kb], b1[kb], acc1);
    storeTileA(t0, acc0, r0, hi, ncol, qbuf, kbuf, vbuf, gbuf, b_gate);
    storeTileA(t1, acc1, r0, hi, ncol, qbuf, kbuf, vbuf, gbuf, b_gate);
  }
  // tile 32: pair bias (only cols 0..3 valid), f32 output
  {
    v8f acc = {};
#pragma unroll
    for (int kb = 0; kb < 4; ++kb) acc = wmma32(afr[kb], b0[kb], acc);
    if (ncol < 4) {
#pragma unroll
      for (int v = 0; v < 8; ++v)
        biasbuf[(size_t)(r0 + v + 8 * hi) * 4 + ncol] = acc[v];
    }
  }
}

// ---------------- kernel B: row-wise attention per (r, h) ----------------
__global__ __launch_bounds__(128) void kB(const half_t* __restrict__ qbuf,
                                          const half_t* __restrict__ kbuf,
                                          const half_t* __restrict__ vbuf,
                                          const half_t* __restrict__ gbuf,
                                          const float* __restrict__ biasbuf,
                                          half_t* __restrict__ aobuf) {
  __shared__ half_t Vt[32 * 256];          // V transposed: [d][j]
  __shared__ half_t attnS[4][16 * 256];    // per-wave normalized attention tile
  const int r = blockIdx.x >> 2;
  const int h = blockIdx.x & 3;
  const int tid = threadIdx.x;
  const int w = tid >> 5, lane = tid & 31;
  const int m = lane & 15, hi = lane >> 4;

  for (int idx = tid; idx < 8192; idx += 128) {
    int j = idx >> 5, d = idx & 31;
    Vt[d * 256 + j] = vbuf[(size_t)(r * 256 + j) * 128 + h * 32 + d];
  }
  __syncthreads();

  const float scale = 0.17677669529663687f;  // 1/sqrt(32)
  for (int itl = 0; itl < 4; ++itl) {
    const int it = itl * 4 + w;
    const half_t* qc = qbuf + (size_t)(r * 256 + it * 16 + m) * 128 + h * 32 + hi * 8;
    v16h aQ = frag2(qc, qc + 16);

    v8f lg[16];
#pragma unroll
    for (int jt = 0; jt < 16; ++jt) {
      const half_t* kc = kbuf + (size_t)(r * 256 + jt * 16 + m) * 128 + h * 32 + hi * 8;
      v16h bK = frag2(kc, kc + 16);
      v8f c = {};
      c = wmma32(aQ, bK, c);
#pragma unroll
      for (int v = 0; v < 8; ++v) {
        int iG = it * 16 + v + 8 * hi;
        int jG = jt * 16 + m;
        lg[jt][v] = c[v] * scale + biasbuf[((size_t)iG * 256 + jG) * 4 + h];
      }
    }
    // softmax over j: 16 tiles in-register + 16-lane cross-lane reduction
    float rmax[8], rsum[8];
#pragma unroll
    for (int v = 0; v < 8; ++v) {
      float mx = lg[0][v];
#pragma unroll
      for (int jt = 1; jt < 16; ++jt) mx = fmaxf(mx, lg[jt][v]);
      mx = fmaxf(mx, __shfl_xor(mx, 1, 32));
      mx = fmaxf(mx, __shfl_xor(mx, 2, 32));
      mx = fmaxf(mx, __shfl_xor(mx, 4, 32));
      mx = fmaxf(mx, __shfl_xor(mx, 8, 32));
      rmax[v] = mx;
    }
#pragma unroll
    for (int v = 0; v < 8; ++v) {
      float s = 0.f;
#pragma unroll
      for (int jt = 0; jt < 16; ++jt) { float e = __expf(lg[jt][v] - rmax[v]); lg[jt][v] = e; s += e; }
      s += __shfl_xor(s, 1, 32);
      s += __shfl_xor(s, 2, 32);
      s += __shfl_xor(s, 4, 32);
      s += __shfl_xor(s, 8, 32);
      rsum[v] = __builtin_amdgcn_rcpf(s);
    }
    half_t* aw = attnS[w];
#pragma unroll
    for (int jt = 0; jt < 16; ++jt)
#pragma unroll
      for (int v = 0; v < 8; ++v)
        aw[(v + 8 * hi) * 256 + jt * 16 + m] = (half_t)(lg[jt][v] * rsum[v]);
    asm volatile("s_wait_dscnt 0" ::: "memory");  // same-wave LDS RAW across lanes

    // attn @ V -> [16 x 32], apply gate, store f16
#pragma unroll
    for (int dt = 0; dt < 2; ++dt) {
      v8f acc = {};
#pragma unroll
      for (int jb = 0; jb < 8; ++jb) {
        const half_t* ac = aw + m * 256 + jb * 32 + hi * 8;
        v16h aA = frag2(ac, ac + 16);
        const half_t* vc = Vt + (dt * 16 + m) * 256 + jb * 32 + hi * 8;
        v16h bV = frag2(vc, vc + 16);
        acc = wmma32(aA, bV, acc);
      }
#pragma unroll
      for (int v = 0; v < 8; ++v) {
        size_t row = (size_t)(r * 256 + it * 16 + v + 8 * hi);
        int col = h * 32 + dt * 16 + m;
        float g = (float)gbuf[row * 128 + col];
        aobuf[row * 128 + col] = (half_t)(g * acc[v]);
      }
    }
  }
}

// ---------------- kernel C: output projection + b_out ----------------
__global__ __launch_bounds__(128) void kC(const half_t* __restrict__ aobuf,
                                          const half_t* __restrict__ packedW,
                                          const float* __restrict__ b_out,
                                          float* __restrict__ out) {
  const int w = threadIdx.x >> 5, lane = threadIdx.x & 31;
  const int m = lane & 15, hi = lane >> 4;
  const int r0 = (blockIdx.x * 4 + w) * 16;
  v16h afr[4];
#pragma unroll
  for (int kb = 0; kb < 4; ++kb) {
    const half_t* c0 = aobuf + (size_t)(r0 + m) * 128 + kb * 32 + hi * 8;
    afr[kb] = frag2(c0, c0 + 16);
  }
  v16h b0[4], b1[4];
  loadB4(packedW, 33, lane, b0);
#pragma unroll 1
  for (int tp = 0; tp < 4; ++tp) {
    const int n0 = tp * 2, n1 = tp * 2 + 1;
    loadB4(packedW, 33 + n1, lane, b1);
    v8f acc0 = {};
#pragma unroll
    for (int kb = 0; kb < 4; ++kb) acc0 = wmma32(afr[kb], b0[kb], acc0);
    if (tp < 3) loadB4(packedW, 33 + n1 + 1, lane, b0);  // prefetch next pair
    v8f acc1 = {};
#pragma unroll
    for (int kb = 0; kb < 4; ++kb) acc1 = wmma32(afr[kb], b1[kb], acc1);
    int col0 = n0 * 16 + m, col1 = n1 * 16 + m;
    float bo0 = b_out[col0], bo1 = b_out[col1];
#pragma unroll
    for (int v = 0; v < 8; ++v) {
      out[(size_t)(r0 + v + 8 * hi) * 128 + col0] = acc0[v] + bo0;
      out[(size_t)(r0 + v + 8 * hi) * 128 + col1] = acc1[v] + bo1;
    }
  }
}

extern "C" void kernel_launch(void* const* d_in, const int* in_sizes, int n_in,
                              void* d_out, int out_size, void* d_ws, size_t ws_size,
                              hipStream_t stream) {
  const float* z      = (const float*)d_in[0];
  const float* gamma  = (const float*)d_in[1];
  const float* beta   = (const float*)d_in[2];
  const float* w_qkv  = (const float*)d_in[3];
  const float* w_pair = (const float*)d_in[4];
  const float* w_gate = (const float*)d_in[5];
  const float* b_gate = (const float*)d_in[6];
  const float* w_out  = (const float*)d_in[7];
  const float* b_out  = (const float*)d_in[8];

  char* ws = (char*)d_ws;
  half_t* packedW = (half_t*)(ws + OFF_W);
  half_t* qbuf    = (half_t*)(ws + OFF_Q);
  half_t* kbuf    = (half_t*)(ws + OFF_K);
  half_t* vbuf    = (half_t*)(ws + OFF_V);
  half_t* gbuf    = (half_t*)(ws + OFF_G);
  half_t* aobuf   = (half_t*)(ws + OFF_AO);
  float*  biasbuf = (float*)(ws + OFF_B);

  kP<<<41, 128, 0, stream>>>(w_qkv, w_pair, w_gate, w_out, packedW);
  kA<<<N_ROWS / 64, 128, 0, stream>>>(z, gamma, beta, b_gate, packedW,
                                      qbuf, kbuf, vbuf, gbuf, biasbuf);
  kB<<<L_DIM * H_HEADS, 128, 0, stream>>>(qbuf, kbuf, vbuf, gbuf, biasbuf, aobuf);
  kC<<<N_ROWS / 64, 128, 0, stream>>>(aobuf, packedW, b_out, (float*)d_out);
}
